// GroupQueryAttention_83605833384187
// MI455X (gfx1250) — matrix operand order repaired and checked by reading, not compiled, not run on hardware
//
#include <hip/hip_runtime.h>
#include <math.h>

// ---------------- types ----------------
typedef __bf16 bf16;
typedef bf16  v16bf __attribute__((ext_vector_type(16)));
typedef float v8f   __attribute__((ext_vector_type(8)));

#define WMMA_BF16(a, b, c) \
  __builtin_amdgcn_wmma_f32_16x16x32_bf16(false, (a), false, (b), (short)0, (c), false, false)

// ---------------- problem constants ----------------
#define B_  2
#define T_  2048
#define D_  2048
#define HQ  32
#define HKV 8
#define DH  64
#define SCALE 0.125f   // 1/sqrt(64)

// ---------------- workspace offsets (bytes) ----------------
#define OFF_XB  0u          // x bf16        [4096][2048]        16,777,216
#define OFF_WQT 16777216u   // Wq^T bf16     [2048][2048]         8,388,608
#define OFF_WKT 25165824u   // Wk^T bf16     [512][2048]          2,097,152
#define OFF_WVT 27262976u   // Wv^T bf16     [512][2048]          2,097,152
#define OFF_WOT 29360128u   // Wo^T bf16     [2048][2048]         8,388,608
#define OFF_QB  37748736u   // Q bf16        [B][HQ][T][64]      16,777,216
#define OFF_KB  54525952u   // K bf16        [B][HKV][T][64]      4,194,304
#define OFF_VT  58720256u   // V^T bf16      [B][HKV][64][T]      4,194,304
#define OFF_AT  62914560u   // attn-out bf16 [4096][2048]        16,777,216

// ---------------- WMMA fragment loaders ----------------
// A matrix 16x32 bf16 (ISA 7.12.2): lane m=lane&15, hi=lane>>4.
// VGPR j holds K = 16*(j>>2) + 8*hi + 2*(j&3), +1  (two contiguous bf16 -> one dword).
static __device__ __forceinline__ v16bf ld_frag_a(const bf16* p, int ld, int row0, int kb, int lane) {
  int m  = row0 + (lane & 15);
  int hi = lane >> 4;
  const bf16* q = p + (size_t)m * ld + kb;
  union { v16bf v; unsigned u[8]; } f;
#pragma unroll
  for (int j = 0; j < 8; ++j) {
    int k0 = ((j >> 2) << 4) + (hi << 3) + ((j & 3) << 1);
    f.u[j] = *(const unsigned*)(q + k0);
  }
  return f.v;
}

// B matrix 32x16 bf16, supplied as Bt = B^T stored N x K row-major.
// Lane n=lane&15; VGPR j holds K = 16*(j>>2) + 8*hi + 2*(j&3), +1 (contiguous in Bt row n).
static __device__ __forceinline__ v16bf ld_frag_b(const bf16* p, int ld, int n0, int kb, int lane) {
  int n  = n0 + (lane & 15);
  int hi = lane >> 4;
  const bf16* q = p + (size_t)n * ld + kb;
  union { v16bf v; unsigned u[8]; } f;
#pragma unroll
  for (int j = 0; j < 8; ++j) {
    int k0 = ((j >> 2) << 4) + (hi << 3) + ((j & 3) << 1);
    f.u[j] = *(const unsigned*)(q + k0);
  }
  return f.v;
}

// ---------------- conversion kernels ----------------
__global__ void __launch_bounds__(256) cvt_bf16(const float* __restrict__ src,
                                                bf16* __restrict__ dst, int n) {
  int i = blockIdx.x * 256 + threadIdx.x;
  if (i < n) dst[i] = (bf16)src[i];
}

// src: K x N f32 row-major  ->  dst: N x K bf16 row-major (transpose)
__global__ void __launch_bounds__(256) cvt_transpose(const float* __restrict__ src,
                                                     bf16* __restrict__ dst, int K, int N) {
  int i = blockIdx.x * 256 + threadIdx.x;
  if (i >= N * K) return;
  int nn = i / K, kk = i - nn * K;
  dst[i] = (bf16)src[(size_t)kk * N + nn];
}

// ---------------- register-blocked bf16 WMMA GEMM ----------------
// Each wave owns a 64x64 output block = 4x4 WMMA tiles.
// Per 32-deep K-step: 8 fragment loads (16x b128) feed 16 WMMA -> 1 b128/WMMA.
// C = A(MxK bf16) * Bt^T(Bt is NxK bf16) + bias
// mode 0: bf16 out, per-head layout [B][H][T][64]  (M = B*T, N = H*64)
// mode 1: bf16 out, transposed per-head [B][H][64][T]
// mode 3: f32 out, row-major [M][N]
__global__ void __launch_bounds__(256) gemm64(const bf16* __restrict__ A,
                                              const bf16* __restrict__ Bt,
                                              const float* __restrict__ bias,
                                              void* __restrict__ out,
                                              int M, int N, int K, int mode, int H) {
  int lane  = threadIdx.x & 31;
  int wid   = (blockIdx.x * blockDim.x + threadIdx.x) >> 5;
  int ntile = N >> 6;                       // 64-wide column blocks
  int tm = wid / ntile;
  int tn = wid - tm * ntile;
  if (tm * 64 >= M) return;
  int m0 = tm * 64, n0 = tn * 64;

  v8f acc[4][4];
#pragma unroll
  for (int i = 0; i < 4; ++i)
#pragma unroll
    for (int j = 0; j < 4; ++j) acc[i][j] = (v8f){};

  for (int kb = 0; kb < K; kb += 32) {
    v16bf af[4], bf[4];
#pragma unroll
    for (int i = 0; i < 4; ++i) af[i] = ld_frag_a(A,  K, m0 + i * 16, kb, lane);
#pragma unroll
    for (int j = 0; j < 4; ++j) bf[j] = ld_frag_b(Bt, K, n0 + j * 16, kb, lane);
#pragma unroll
    for (int i = 0; i < 4; ++i)
#pragma unroll
      for (int j = 0; j < 4; ++j)
        acc[i][j] = WMMA_BF16(af[i], bf[j], acc[i][j]);
  }

  int c = lane & 15, hi = lane >> 4;
#pragma unroll
  for (int j = 0; j < 4; ++j) {
    int n = n0 + j * 16 + c;
    float bv = bias[n];
#pragma unroll
    for (int i = 0; i < 4; ++i) {
#pragma unroll
      for (int r = 0; r < 8; ++r) {
        int m = m0 + i * 16 + r + (hi << 3);
        float v = acc[i][j][r] + bv;
        if (mode == 0) {
          int bb = m >> 11, t = m & (T_ - 1);       // T_ == 2048
          int hh = n >> 6,  d = n & 63;
          ((bf16*)out)[(((size_t)(bb * H + hh) * T_ + t) << 6) + d] = (bf16)v;
        } else if (mode == 1) {
          int bb = m >> 11, t = m & (T_ - 1);
          int hh = n >> 6,  d = n & 63;
          ((bf16*)out)[((size_t)(bb * H + hh) * 64 + d) * T_ + t] = (bf16)v;
        } else {
          ((float*)out)[(size_t)m * N + n] = v;
        }
      }
    }
  }
}

// ---------------- flash attention (causal, GQA) ----------------
// One wave owns 16 query rows of one head; loops over 32-key tiles.
__global__ void __launch_bounds__(256) fa_kernel(const bf16* __restrict__ qb,
                                                 const bf16* __restrict__ kbuf,
                                                 const bf16* __restrict__ vtb,
                                                 bf16* __restrict__ ab) {
  __shared__ bf16 plds[8 * 16 * 32];     // 16x32 bf16 P-tile per wave, 8 waves
  int lane = threadIdx.x & 31;
  int wloc = threadIdx.x >> 5;
  int wid  = blockIdx.x * 8 + wloc;
  int qt  = wid % (T_ / 16);
  int h   = (wid / (T_ / 16)) % HQ;
  int b   = wid / ((T_ / 16) * HQ);
  int kvh = h / (HQ / HKV);

  const bf16* qp = qb   + (size_t)(b * HQ  + h)   * T_ * DH;
  const bf16* kp = kbuf + (size_t)(b * HKV + kvh) * T_ * DH;
  const bf16* vp = vtb  + (size_t)(b * HKV + kvh) * DH * T_;   // dim-major [64][T]
  bf16* pl = plds + wloc * (16 * 32);

  int c = lane & 15, hi = lane >> 4;
  int q0 = qt * 16;

  v16bf aq0 = ld_frag_a(qp, DH, q0, 0,  lane);
  v16bf aq1 = ld_frag_a(qp, DH, q0, 32, lane);

  v8f o0 = {}, o1 = {}, o2 = {}, o3 = {};
  float mrow[8], lrow[8];
#pragma unroll
  for (int r = 0; r < 8; ++r) { mrow[r] = -INFINITY; lrow[r] = 0.f; }

  int ktmax = (q0 + 15) >> 5;
  for (int kt = 0; kt <= ktmax; ++kt) {
    int kbase = kt << 5;
    // S = Q * K^T : two 16x16 score tiles (keys kbase..+15, kbase+16..+31)
    v8f s0 = {}, s1 = {};
    s0 = WMMA_BF16(aq0, ld_frag_b(kp, DH, kbase,      0,  lane), s0);
    s0 = WMMA_BF16(aq1, ld_frag_b(kp, DH, kbase,      32, lane), s0);
    s1 = WMMA_BF16(aq0, ld_frag_b(kp, DH, kbase + 16, 0,  lane), s1);
    s1 = WMMA_BF16(aq1, ld_frag_b(kp, DH, kbase + 16, 32, lane), s1);

#pragma unroll
    for (int r = 0; r < 8; ++r) {
      int qg = q0 + r + (hi << 3);
      float x0 = s0[r] * SCALE; if (kbase + c > qg)      x0 = -1e30f;
      float x1 = s1[r] * SCALE; if (kbase + 16 + c > qg) x1 = -1e30f;
      // row max across 16 lanes of this half-wave
      float tm = fmaxf(x0, x1);
      tm = fmaxf(tm, __shfl_xor(tm, 1));
      tm = fmaxf(tm, __shfl_xor(tm, 2));
      tm = fmaxf(tm, __shfl_xor(tm, 4));
      tm = fmaxf(tm, __shfl_xor(tm, 8));
      float mn = fmaxf(mrow[r], tm);
      float al = __expf(mrow[r] - mn);
      float p0 = __expf(x0 - mn);
      float p1 = __expf(x1 - mn);
      float rs = p0 + p1;
      rs += __shfl_xor(rs, 1);
      rs += __shfl_xor(rs, 2);
      rs += __shfl_xor(rs, 4);
      rs += __shfl_xor(rs, 8);
      lrow[r] = lrow[r] * al + rs;
      mrow[r] = mn;
      o0[r] *= al; o1[r] *= al; o2[r] *= al; o3[r] *= al;
      // stage P (C-layout) into LDS as a row-major 16x32 bf16 tile
      int row = r + (hi << 3);
      pl[row * 32 + c]      = (bf16)p0;
      pl[row * 32 + 16 + c] = (bf16)p1;
    }
    asm volatile("s_wait_dscnt 0" ::: "memory");   // order LDS stores vs re-gather
    v16bf ap = ld_frag_a(pl, 32, 0, 0, lane);      // P as A-fragment (16x32)
    // O += P * V  (V^T is dim-major, so B-fragment loads are contiguous dwords)
    o0 = WMMA_BF16(ap, ld_frag_b(vp, T_, 0,  kbase, lane), o0);
    o1 = WMMA_BF16(ap, ld_frag_b(vp, T_, 16, kbase, lane), o1);
    o2 = WMMA_BF16(ap, ld_frag_b(vp, T_, 32, kbase, lane), o2);
    o3 = WMMA_BF16(ap, ld_frag_b(vp, T_, 48, kbase, lane), o3);
  }

#pragma unroll
  for (int r = 0; r < 8; ++r) {
    float inv = 1.f / lrow[r];
    size_t rowg = (size_t)(b * T_ + q0 + r + (hi << 3)) * D_;
    int colb = h * DH;
    ab[rowg + colb + c]      = (bf16)(o0[r] * inv);
    ab[rowg + colb + 16 + c] = (bf16)(o1[r] * inv);
    ab[rowg + colb + 32 + c] = (bf16)(o2[r] * inv);
    ab[rowg + colb + 48 + c] = (bf16)(o3[r] * inv);
  }
}

// ---------------- host launcher ----------------
extern "C" void kernel_launch(void* const* d_in, const int* in_sizes, int n_in,
                              void* d_out, int out_size, void* d_ws, size_t ws_size,
                              hipStream_t stream) {
  const float* x  = (const float*)d_in[0];
  const float* Wq = (const float*)d_in[1];
  const float* bq = (const float*)d_in[2];
  const float* Wk = (const float*)d_in[3];
  const float* bk = (const float*)d_in[4];
  const float* Wv = (const float*)d_in[5];
  const float* bv = (const float*)d_in[6];
  const float* Wo = (const float*)d_in[7];
  const float* bo = (const float*)d_in[8];

  char* ws = (char*)d_ws;
  bf16* xb  = (bf16*)(ws + OFF_XB);
  bf16* wqt = (bf16*)(ws + OFF_WQT);
  bf16* wkt = (bf16*)(ws + OFF_WKT);
  bf16* wvt = (bf16*)(ws + OFF_WVT);
  bf16* wot = (bf16*)(ws + OFF_WOT);
  bf16* qbf = (bf16*)(ws + OFF_QB);
  bf16* kbf = (bf16*)(ws + OFF_KB);
  bf16* vtb = (bf16*)(ws + OFF_VT);
  bf16* atb = (bf16*)(ws + OFF_AT);

  const int M = B_ * T_;                 // 4096

  // 1) precision conversion / weight transposition
  cvt_bf16<<<(M * D_) / 256, 256, 0, stream>>>(x, xb, M * D_);
  cvt_transpose<<<(D_ * D_) / 256, 256, 0, stream>>>(Wq, wqt, D_, D_);
  cvt_transpose<<<(D_ * 512) / 256, 256, 0, stream>>>(Wk, wkt, D_, 512);
  cvt_transpose<<<(D_ * 512) / 256, 256, 0, stream>>>(Wv, wvt, D_, 512);
  cvt_transpose<<<(D_ * D_) / 256, 256, 0, stream>>>(Wo, wot, D_, D_);

  // 2) projections (WMMA bf16, f32 accumulate), 64x64 per wave
  gemm64<<<((M / 64) * (D_ / 64)) / 8, 256, 0, stream>>>(xb, wqt, bq, qbf, M, D_, D_, 0, HQ);
  gemm64<<<((M / 64) * (512 / 64)) / 8, 256, 0, stream>>>(xb, wkt, bk, kbf, M, 512, D_, 0, HKV);
  gemm64<<<((M / 64) * (512 / 64)) / 8, 256, 0, stream>>>(xb, wvt, bv, vtb, M, 512, D_, 1, HKV);

  // 3) causal GQA flash attention
  fa_kernel<<<(B_ * HQ * (T_ / 16)) / 8, 256, 0, stream>>>(qbf, kbf, vtb, atb);

  // 4) output projection -> f32 out + bias
  gemm64<<<((M / 64) * (D_ / 64)) / 8, 256, 0, stream>>>(atb, wot, bo, d_out, M, D_, D_, 3, 0);
}
